// CrossModalAttention_89137751261863
// MI455X (gfx1250) — matrix operand order repaired and checked by reading, not compiled, not run on hardware
//
#include <hip/hip_runtime.h>

typedef _Float16 half16 __attribute__((ext_vector_type(16)));
typedef _Float16 half8  __attribute__((ext_vector_type(8)));
typedef float    f32x8  __attribute__((ext_vector_type(8)));
typedef float    f32x4  __attribute__((ext_vector_type(4)));

#define NH 4
#define HD 16
#define AD 64
#define CC 256
#define SS 1024
#define BB 16
#define ATT_SCALE 0.25f

// workspace layout, offsets in _Float16 elements
#define Q_OFF   ((size_t)0)         // [2][B][NH][S][HD]
#define K_OFF   ((size_t)2097152)   // [2][B][NH][S][HD]
#define V_OFF   ((size_t)4194304)   // [2][B][NH][HD][S]  (transposed)
#define ATT_OFF ((size_t)6291456)   // [2][B][S][AD]
#define WQ_OFF  ((size_t)8388608)   // [2][192][256] f16 (transposed qkv weights)
#define WO_OFF  ((size_t)8486912)   // [2][256][64]  f16 (transposed out weights)

union H16 { half16 v; half8 h[2]; };

__device__ __forceinline__ f32x8 wmma16(half16 a, half16 b, f32x8 c) {
  return __builtin_amdgcn_wmma_f32_16x16x32_f16(false, a, false, b, (short)0, c,
                                                false, false);
}

// ---------------------------------------------------------------------------
// Kernel 0: convert + transpose weights to f16 (L2-resident afterwards)
// ---------------------------------------------------------------------------
__global__ void prep_weights(const float* __restrict__ cqkv_w,
                             const float* __restrict__ bqkv_w,
                             const float* __restrict__ cout_w,
                             const float* __restrict__ bout_w,
                             _Float16* __restrict__ ws) {
  int idx = blockIdx.x * blockDim.x + threadIdx.x;
  const int NQ = 2 * 192 * CC;
  if (idx < NQ) {
    int mod = idx / (192 * CC);
    int r = idx - mod * 192 * CC;
    int j = r / CC, c = r % CC;                 // wqkvT[mod][j][c] = w[c][j]
    const float* w = mod ? bqkv_w : cqkv_w;
    ws[WQ_OFF + idx] = (_Float16)w[c * 192 + j];
  } else {
    int k = idx - NQ;
    if (k < 2 * CC * AD) {
      int mod = k / (CC * AD);
      int r = k - mod * CC * AD;
      int c = r / AD, a = r % AD;               // woutT[mod][c][a] = w[a][c]
      const float* w = mod ? bout_w : cout_w;
      ws[WO_OFF + k] = (_Float16)w[a * CC + c];
    }
  }
}

// ---------------------------------------------------------------------------
// Kernel 1: QKV projection. One 16-row s-tile per block (4 waves).
// Each wave owns 3 N-tiles of 16 cols (192 total). K-loop: 8 steps of 32.
// ---------------------------------------------------------------------------
__global__ __launch_bounds__(128) void qkv_kernel(const float* __restrict__ color,
                                                  const float* __restrict__ bright,
                                                  const float* __restrict__ cqkv_b,
                                                  const float* __restrict__ bqkv_b,
                                                  _Float16* __restrict__ ws) {
  int x = blockIdx.x;
  int stile = x & 63;
  int b = (x >> 6) & 15;
  int mod = x >> 10;
  int s0 = stile * 16;

  const float* in = (mod ? bright : color) + (size_t)b * CC * SS;
  const float* bias = mod ? bqkv_b : cqkv_b;

  __shared__ _Float16 xt[16][CC + 8];

  int tid = threadIdx.x;
  for (int idx = tid; idx < 16 * CC; idx += 128) {
    int si = idx & 15, c = idx >> 4;
    xt[si][c] = (_Float16)in[(size_t)c * SS + s0 + si];
  }
  __syncthreads();

  int wave = tid >> 5, lane = tid & 31;
  int m = lane & 15, hi = lane >> 4;
  const _Float16* wq = ws + WQ_OFF + (size_t)mod * (192 * CC);

  f32x8 zf = {};
  f32x8 acc[3];
  for (int t = 0; t < 3; ++t) acc[t] = zf;

  for (int k = 0; k < CC; k += 32) {
    H16 A;
    const _Float16* ar = &xt[m][k + hi * 8];
    A.h[0] = *(const half8*)ar;
    A.h[1] = *(const half8*)(ar + 16);
    for (int t = 0; t < 3; ++t) {
      int j = (wave * 3 + t) * 16 + m;          // B-col n == lane&15
      const _Float16* br = wq + (size_t)j * CC + k + hi * 8;
      H16 Bf;
      Bf.h[0] = *(const half8*)br;
      Bf.h[1] = *(const half8*)(br + 16);
      acc[t] = wmma16(A.v, Bf.v, acc[t]);
    }
  }

  for (int t = 0; t < 3; ++t) {
    int j = (wave * 3 + t) * 16 + m;
    int which = j >> 6, h = (j >> 4) & 3, d = j & 15;
    float bv = bias[j];
    size_t head = (((size_t)mod * BB + b) * NH + h);
    if (which == 0) {
      _Float16* qp = ws + Q_OFF + head * SS * HD;
      for (int r = 0; r < 8; ++r)
        qp[(size_t)(s0 + hi * 8 + r) * HD + d] = (_Float16)(acc[t][r] + bv);
    } else if (which == 1) {
      _Float16* kp = ws + K_OFF + head * SS * HD;
      for (int r = 0; r < 8; ++r)
        kp[(size_t)(s0 + hi * 8 + r) * HD + d] = (_Float16)(acc[t][r] + bv);
    } else {
      half8 vv;
      for (int r = 0; r < 8; ++r) vv[r] = (_Float16)(acc[t][r] + bv);
      *(half8*)(ws + V_OFF + (head * HD + d) * SS + s0 + hi * 8) = vv;  // V^T
    }
  }
}

// ---------------------------------------------------------------------------
// Kernel 2: flash attention, one wave per (mod, b, head, q-tile).
// Q of modality `mod` attends to K/V of modality `mod^1`.
// ---------------------------------------------------------------------------
__global__ __launch_bounds__(32) void attn_kernel(_Float16* __restrict__ ws) {
  int x = blockIdx.x;
  int qt = x & 63;
  int h = (x >> 6) & 3;
  int b = (x >> 8) & 15;
  int mod = x >> 12;
  int sq0 = qt * 16;

  int lane = threadIdx.x;
  int m = lane & 15, hi = lane >> 4;

  size_t qhead = ((size_t)mod * BB + b) * NH + h;
  size_t khead = ((size_t)(mod ^ 1) * BB + b) * NH + h;

  __shared__ float sc[16][33];
  __shared__ _Float16 pf[16][40];
  __shared__ float rmax[16], rsum[16], corr[16];

  H16 A;   // Q tile: K-dim 16 padded to 32 (upper half zero)
  {
    const _Float16* qp = ws + Q_OFF + (qhead * SS + sq0 + m) * HD + hi * 8;
    A.h[0] = *(const half8*)qp;
    half8 z;
    for (int i = 0; i < 8; ++i) z[i] = (_Float16)0.f;
    A.h[1] = z;
  }
  if (lane < 16) { rmax[lane] = -1e30f; rsum[lane] = 0.f; }
  __syncthreads();

  f32x8 O = {};
  f32x8 zf = {};

  const _Float16* kbase = ws + K_OFF + khead * SS * HD;
  const _Float16* vbase = ws + V_OFF + khead * HD * SS;

  for (int kb = 0; kb < SS; kb += 32) {
    // two 16x16 score tiles via zero-padded K-dim
    for (int hf = 0; hf < 2; ++hf) {
      H16 Bf;
      {                                           // elements 8..15: K-dim 16..31 -> 0
        const _Float16* kp = kbase + (size_t)(kb + hf * 16 + m) * HD + hi * 8;
        Bf.h[0] = *(const half8*)kp;
        for (int i = 0; i < 8; ++i) Bf.h[1][i] = (_Float16)0.f;
      }
      f32x8 S = wmma16(A.v, Bf.v, zf);
      for (int r = 0; r < 8; ++r)
        sc[hi * 8 + r][hf * 16 + m] = S[r] * ATT_SCALE;
    }
    __syncthreads();

    // online softmax, one row per lane 0..15
    if (lane < 16) {
      float mo = rmax[lane];
      float mb = mo;
      for (int j = 0; j < 32; ++j) mb = fmaxf(mb, sc[lane][j]);
      float c = __expf(mo - mb);
      float s = 0.f;
      for (int j = 0; j < 32; ++j) {
        float p = __expf(sc[lane][j] - mb);
        pf[lane][j] = (_Float16)p;
        s += p;
      }
      rsum[lane] = rsum[lane] * c + s;
      rmax[lane] = mb;
      corr[lane] = c;
    }
    __syncthreads();

    for (int r = 0; r < 8; ++r) O[r] *= corr[hi * 8 + r];

    H16 P;                                        // 16x32 probs, full K=32
    P.h[0] = *(const half8*)&pf[m][hi * 8];
    P.h[1] = *(const half8*)&pf[m][16 + hi * 8];
    H16 V;                                        // V^T rows are contiguous in s
    const _Float16* vp = vbase + (size_t)m * SS + kb + hi * 8;
    V.h[0] = *(const half8*)vp;
    V.h[1] = *(const half8*)(vp + 16);
    O = wmma16(P.v, V.v, O);
    __syncthreads();
  }

  _Float16* ap = ws + ATT_OFF + (((size_t)mod * BB + b) * SS + sq0) * AD + h * HD + m;
  for (int r = 0; r < 8; ++r) {
    float o = O[r] / rsum[hi * 8 + r];
    ap[(size_t)(hi * 8 + r) * AD] = (_Float16)o;
  }
}

// ---------------------------------------------------------------------------
// Kernel 3: output projection + bias + residual, back to (B,C,H,W) f32.
// One 16-row s-tile per block; each wave owns 4 N-tiles (64 of 256 cols).
// ---------------------------------------------------------------------------
__global__ __launch_bounds__(128) void out_kernel(const float* __restrict__ color,
                                                  const float* __restrict__ bright,
                                                  const float* __restrict__ cout_b,
                                                  const float* __restrict__ bout_b,
                                                  const _Float16* __restrict__ ws,
                                                  float* __restrict__ out) {
  int x = blockIdx.x;
  int stile = x & 63;
  int b = (x >> 6) & 15;
  int mod = x >> 10;
  int s0 = stile * 16;
  int tid = threadIdx.x;

  __shared__ _Float16 at[16][AD + 8];
  const _Float16* ab = ws + ATT_OFF + (((size_t)mod * BB + b) * SS + s0) * AD;
  for (int idx = tid; idx < 16 * AD; idx += 128) {
    int si = idx >> 6, a = idx & 63;
    at[si][a] = ab[(size_t)si * AD + a];
  }
  __syncthreads();

  int wave = tid >> 5, lane = tid & 31;
  int m = lane & 15, hi = lane >> 4;
  const _Float16* wo = ws + WO_OFF + (size_t)mod * (CC * AD);
  const float* bias = mod ? bout_b : cout_b;
  const float* res = (mod ? bright : color) + (size_t)b * CC * SS;
  float* op = out + ((size_t)mod * BB + b) * CC * SS;

  f32x8 zf = {};
  f32x8 acc[4];
  for (int t = 0; t < 4; ++t) acc[t] = zf;

  for (int k = 0; k < AD; k += 32) {
    H16 A;
    A.h[0] = *(const half8*)&at[m][k + hi * 8];
    A.h[1] = *(const half8*)&at[m][k + 16 + hi * 8];
    for (int t = 0; t < 4; ++t) {
      int c = (wave * 4 + t) * 16 + m;
      const _Float16* br = wo + (size_t)c * AD + k + hi * 8;
      H16 Bf;
      Bf.h[0] = *(const half8*)br;
      Bf.h[1] = *(const half8*)(br + 16);
      acc[t] = wmma16(A.v, Bf.v, acc[t]);
    }
  }

  for (int t = 0; t < 4; ++t) {
    int c = (wave * 4 + t) * 16 + m;
    float bv = bias[c];
    size_t off = (size_t)c * SS + s0 + hi * 8;
    f32x4 r0 = *(const f32x4*)(res + off);
    f32x4 r1 = *(const f32x4*)(res + off + 4);
    f32x4 o0, o1;
    for (int r = 0; r < 4; ++r) {
      o0[r] = acc[t][r] + bv + r0[r];
      o1[r] = acc[t][r + 4] + bv + r1[r];
    }
    *(f32x4*)(op + off) = o0;
    *(f32x4*)(op + off + 4) = o1;
  }
}

// ---------------------------------------------------------------------------
extern "C" void kernel_launch(void* const* d_in, const int* in_sizes, int n_in,
                              void* d_out, int out_size, void* d_ws, size_t ws_size,
                              hipStream_t stream) {
  const float* color  = (const float*)d_in[0];
  const float* bright = (const float*)d_in[1];
  const float* cqkv_w = (const float*)d_in[2];
  const float* cqkv_b = (const float*)d_in[3];
  const float* bqkv_w = (const float*)d_in[4];
  const float* bqkv_b = (const float*)d_in[5];
  const float* cout_w = (const float*)d_in[6];
  const float* cout_b = (const float*)d_in[7];
  const float* bout_w = (const float*)d_in[8];
  const float* bout_b = (const float*)d_in[9];
  float* out = (float*)d_out;
  _Float16* ws = (_Float16*)d_ws;

  prep_weights<<<512, 256, 0, stream>>>(cqkv_w, bqkv_w, cout_w, bout_w, ws);
  qkv_kernel<<<2048, 128, 0, stream>>>(color, bright, cqkv_b, bqkv_b, ws);
  attn_kernel<<<8192, 32, 0, stream>>>(ws);
  out_kernel<<<2048, 128, 0, stream>>>(color, bright, cout_b, bout_b, ws, out);
}
